// PointNet2Encoder_86346022518891
// MI455X (gfx1250) — matrix operand order repaired and checked by reading, not compiled, not run on hardware
//
#include <hip/hip_runtime.h>
#include <hip/hip_bf16.h>
#include <cstdint>
#include <cstddef>

typedef __bf16 bf16;
typedef __attribute__((ext_vector_type(16))) __bf16 bf16x16;
typedef __attribute__((ext_vector_type(8)))  __bf16 bf16x8;
typedef __attribute__((ext_vector_type(8)))  float  f32x8;

#define BN_FOLD 0.9999950000374997f   /* 1/sqrt(1+1e-5) */
#define AGENT __HIP_MEMORY_SCOPE_AGENT

static inline long cdivl(long a, long b) { return (a + b - 1) / b; }
static inline int ceil32(int x) { return (x + 31) & ~31; }

// ---------------------------------------------------------------------------
// extract: data[N,9] f32 -> xyz[N,3] f32 packed + feat[N,9] bf16
// ---------------------------------------------------------------------------
__global__ void extract_kernel(const float* __restrict__ data, int n,
                               float* __restrict__ xyz, bf16* __restrict__ feat) {
  int i = blockIdx.x * blockDim.x + threadIdx.x;
  if (i >= n) return;
  const float* p = data + (size_t)i * 9;
  xyz[3 * i + 0] = p[0];
  xyz[3 * i + 1] = p[1];
  xyz[3 * i + 2] = p[2];
  bf16* f = feat + (size_t)i * 9;
#pragma unroll
  for (int c = 0; c < 9; c++) f[c] = (bf16)p[c];
}

// ---------------------------------------------------------------------------
// Multi-workgroup farthest point sampling with persistent-grid sync.
// Each block keeps its slice (xyz + running min-dist) fully in LDS
// (CDNA5: 320KB LDS per WGP). Cross-block argmax via agent-scope atomics:
//   arrive counter (monotonic, target (s+1)*G), generation counter `gen`
//   release-published by block 0 together with next centroid coords.
// ---------------------------------------------------------------------------
#define FPS_T 256
#define FPS_SLICE 8192

__global__ void fps_init_kernel(int* __restrict__ sync2) {
  sync2[0] = 0;  // arrive
  sync2[1] = 0;  // gen
}

__global__ __launch_bounds__(FPS_T) void fps_multi_kernel(
    const float* __restrict__ xyz, int N, int npoint, int G,
    int* __restrict__ out_idx, float* __restrict__ pval, int* __restrict__ pidx,
    int* __restrict__ sync2, float* __restrict__ gxyz) {
  __shared__ float sx[FPS_SLICE], sy[FPS_SLICE], sz[FPS_SLICE], sd[FPS_SLICE];
  __shared__ float rval[FPS_T];
  __shared__ int   ridx[FPS_T];
  __shared__ float fco[3];

  int b = blockIdx.x, t = threadIdx.x;
  int slice = (N + G - 1) / G;
  int beg = b * slice;
  int end = beg + slice; if (end > N) end = N;
  int cnt = end - beg; if (cnt < 0) cnt = 0;

  for (int j = t; j < cnt; j += FPS_T) {
    sx[j] = xyz[3 * (beg + j) + 0];
    sy[j] = xyz[3 * (beg + j) + 1];
    sz[j] = xyz[3 * (beg + j) + 2];
    sd[j] = 1e10f;
  }
  __syncthreads();

  int* arrive = sync2 + 0;
  int* gen    = sync2 + 1;

  for (int s = 0; s < npoint; s++) {
    if (t == 0) {
      if (s == 0) {
        fco[0] = xyz[0]; fco[1] = xyz[1]; fco[2] = xyz[2];
        if (b == 0) out_idx[0] = 0;
      } else {
        while (__hip_atomic_load(gen, __ATOMIC_ACQUIRE, AGENT) < s) { }
        fco[0] = __hip_atomic_load(&gxyz[0], __ATOMIC_RELAXED, AGENT);
        fco[1] = __hip_atomic_load(&gxyz[1], __ATOMIC_RELAXED, AGENT);
        fco[2] = __hip_atomic_load(&gxyz[2], __ATOMIC_RELAXED, AGENT);
      }
    }
    __syncthreads();
    float fx = fco[0], fy = fco[1], fz = fco[2];

    float best = -1e30f;
    int   bi = 0;
    for (int j = t; j < cnt; j += FPS_T) {
      float dx = sx[j] - fx, dy = sy[j] - fy, dz = sz[j] - fz;
      float d = dx * dx + dy * dy + dz * dz;
      float dd = fminf(sd[j], d);
      sd[j] = dd;
      if (dd > best) { best = dd; bi = beg + j; }
    }
    rval[t] = best; ridx[t] = bi;
    __syncthreads();
    for (int o = FPS_T / 2; o > 0; o >>= 1) {
      if (t < o) {
        float ov = rval[t + o]; int oi = ridx[t + o];
        if (ov > rval[t] || (ov == rval[t] && oi < ridx[t])) { rval[t] = ov; ridx[t] = oi; }
      }
      __syncthreads();
    }
    if (t == 0) {
      __hip_atomic_store(&pval[b], rval[0], __ATOMIC_RELAXED, AGENT);
      __hip_atomic_store(&pidx[b], ridx[0], __ATOMIC_RELAXED, AGENT);
      __hip_atomic_fetch_add(arrive, 1, __ATOMIC_RELEASE, AGENT);
      if (b == 0) {
        while (__hip_atomic_load(arrive, __ATOMIC_ACQUIRE, AGENT) < (s + 1) * G) { }
        float bv = -1e30f; int bix = 0;
        for (int q = 0; q < G; q++) {
          float v = __hip_atomic_load(&pval[q], __ATOMIC_RELAXED, AGENT);
          int  ii = __hip_atomic_load(&pidx[q], __ATOMIC_RELAXED, AGENT);
          if (v > bv || (v == bv && ii < bix)) { bv = v; bix = ii; }
        }
        if (s + 1 < npoint) out_idx[s + 1] = bix;
        __hip_atomic_store(&gxyz[0], xyz[3 * bix + 0], __ATOMIC_RELAXED, AGENT);
        __hip_atomic_store(&gxyz[1], xyz[3 * bix + 1], __ATOMIC_RELAXED, AGENT);
        __hip_atomic_store(&gxyz[2], xyz[3 * bix + 2], __ATOMIC_RELAXED, AGENT);
        __hip_atomic_store(gen, s + 1, __ATOMIC_RELEASE, AGENT);
      }
    }
    __syncthreads();
  }
}

// gather xyz[idx[i]] -> out[i]
__global__ void gather_xyz_kernel(const float* __restrict__ xyz, const int* __restrict__ idx,
                                  int n, float* __restrict__ out) {
  int i = blockIdx.x * blockDim.x + threadIdx.x;
  if (i >= n) return;
  int j = idx[i];
  out[3 * i + 0] = xyz[3 * j + 0];
  out[3 * i + 1] = xyz[3 * j + 1];
  out[3 * i + 2] = xyz[3 * j + 2];
}

// ---------------------------------------------------------------------------
// Ball query: one wave32 per centroid; first K in-radius indices (ascending),
// remaining slots filled with the first found index (matches reference).
// ---------------------------------------------------------------------------
__global__ void ball_query_kernel(const float* __restrict__ xyz, int N,
                                  const float* __restrict__ newxyz, int np,
                                  float r2, int K, int* __restrict__ idxout) {
  int wid  = (blockIdx.x * blockDim.x + threadIdx.x) >> 5;
  int lane = threadIdx.x & 31;
  if (wid >= np) return;
  float cx = newxyz[3 * wid + 0], cy = newxyz[3 * wid + 1], cz = newxyz[3 * wid + 2];
  int cnt = 0;
  int first = -1;
  for (int base = 0; base < N && cnt < K; base += 32) {
    int j = base + lane;
    bool in = false;
    if (j < N) {
      float dx = xyz[3 * j + 0] - cx;
      float dy = xyz[3 * j + 1] - cy;
      float dz = xyz[3 * j + 2] - cz;
      in = (dx * dx + dy * dy + dz * dz) <= r2;
    }
    unsigned m = (unsigned)__ballot(in);
    if (first < 0 && m) first = base + __ffs(m) - 1;
    int pos = cnt + __popc(m & ((1u << lane) - 1u));
    if (in && pos < K) idxout[(size_t)wid * K + pos] = j;
    cnt += __popc(m);
  }
  if (first < 0) first = 0;
  for (int p = cnt + lane; p < K; p += 32) idxout[(size_t)wid * K + p] = first;
}

// ---------------------------------------------------------------------------
// Grouped gather+concat: X[row, :] = [feat[idx], xyz[idx]-center, 0-pad] bf16
// ---------------------------------------------------------------------------
__global__ void sa_gather_kernel(const float* __restrict__ xyz, const bf16* __restrict__ feats,
                                 int C, const float* __restrict__ newxyz,
                                 const int* __restrict__ idx, int K, int cinp,
                                 bf16* __restrict__ X) {
  int row = blockIdx.x;
  int g = row / K;
  int j = idx[row];
  float gx = xyz[3 * j + 0] - newxyz[3 * g + 0];
  float gy = xyz[3 * j + 1] - newxyz[3 * g + 1];
  float gz = xyz[3 * j + 2] - newxyz[3 * g + 2];
  bf16* xr = X + (size_t)row * cinp;
  for (int c = threadIdx.x; c < cinp; c += blockDim.x) {
    float v;
    if (c < C)           v = (float)feats[(size_t)j * C + c];
    else if (c == C)     v = gx;
    else if (c == C + 1) v = gy;
    else if (c == C + 2) v = gz;
    else                 v = 0.f;
    xr[c] = (bf16)v;
  }
}

// ---------------------------------------------------------------------------
// Weight packing: fold BN scale into bf16 weights, bias -> f32
// ---------------------------------------------------------------------------
__global__ void pack_w_kernel(const float* __restrict__ w, const float* __restrict__ g,
                              int cout, int cin, int cinp, int coutp,
                              bf16* __restrict__ Wp) {
  int i = blockIdx.x * blockDim.x + threadIdx.x;
  if (i >= coutp * cinp) return;
  int o = i / cinp, c = i - o * cinp;
  float v = 0.f;
  if (o < cout && c < cin) v = w[(size_t)o * cin + c] * (g[o] * BN_FOLD);
  Wp[i] = (bf16)v;
}

__global__ void pack_b_kernel(const float* __restrict__ b, const float* __restrict__ g,
                              const float* __restrict__ be, int cout, int coutp,
                              float* __restrict__ bp) {
  int o = blockIdx.x * blockDim.x + threadIdx.x;
  if (o >= coutp) return;
  bp[o] = (o < cout) ? (b[o] * (g[o] * BN_FOLD) + be[o]) : 0.f;
}

// ---------------------------------------------------------------------------
// WMMA GEMM: Y[M,coutp] = relu(X[M,cinp] @ W[coutp,cinp]^T + bias)
// One wave32 per (16 rows x NT*16 cols) strip: the A fragment is loaded once
// per K-step and reused for NT v_wmma_f32_16x16x32_bf16 ops -> (2/NT + 2)
// b128 loads per WMMA instead of 4, and X is re-read NT x less from L2.
// A layout (16-bit 16x32):  lane l, hi=l/16, row=l%16; elems 0..7 -> K=k+hi*8..,
//                           elems 8..15 -> K=k+16+hi*8..
// B layout (16-bit 32x16):  lane l, col=l%16; same K mapping as A
// D layout (f32 16x16):     VGPR i, lane l -> row = hi*8+i, col = l%16
// ---------------------------------------------------------------------------
template <int NT>
__global__ void wmma_gemm_kernel(const bf16* __restrict__ X, const bf16* __restrict__ W,
                                 const float* __restrict__ bias, int M, int cinp, int coutp,
                                 bf16* __restrict__ outb, float* __restrict__ outf) {
  int wid  = (blockIdx.x * blockDim.x + threadIdx.x) >> 5;
  int lane = threadIdx.x & 31;
  int ng  = (coutp >> 4) / NT;   // column-tile groups
  int ntm = M >> 4;
  int tm = wid / ng;
  int tg = wid - tm * ng;
  if (tm >= ntm) return;   // wave-uniform: EXEC stays all-ones for WMMA
  int hi = lane >> 4;
  int lr = lane & 15;
  const bf16* xrow = X + (size_t)((tm << 4) + lr) * cinp;
  const bf16* wr[NT];
#pragma unroll
  for (int t = 0; t < NT; t++)
    wr[t] = W + (size_t)(((tg * NT + t) << 4) + lr) * cinp;

  f32x8 acc[NT] = {};
  for (int k = 0; k < cinp; k += 32) {
    bf16x8 a0 = *(const bf16x8*)(xrow + k + hi * 8);
    bf16x8 a1 = *(const bf16x8*)(xrow + k + 16 + hi * 8);
    bf16x16 a = __builtin_shufflevector(a0, a1, 0,1,2,3,4,5,6,7,8,9,10,11,12,13,14,15);
#pragma unroll
    for (int t = 0; t < NT; t++) {
      bf16x8 b0 = *(const bf16x8*)(wr[t] + k + hi * 8);
      bf16x8 b1 = *(const bf16x8*)(wr[t] + k + 16 + hi * 8);
      bf16x16 bb = __builtin_shufflevector(b0, b1, 0,1,2,3,4,5,6,7,8,9,10,11,12,13,14,15);
      acc[t] = __builtin_amdgcn_wmma_f32_16x16x32_bf16(false, a, false, bb, (short)0,
                                                       acc[t], false, false);
    }
  }
#pragma unroll
  for (int t = 0; t < NT; t++) {
    int col = ((tg * NT + t) << 4) + lr;
    float bc = bias[col];
#pragma unroll
    for (int i = 0; i < 8; i++) {
      int r = (tm << 4) + hi * 8 + i;
      float v = acc[t][i] + bc;
      v = v > 0.f ? v : 0.f;
      if (outf) outf[(size_t)r * coutp + col] = v;
      else      outb[(size_t)r * coutp + col] = (bf16)v;
    }
  }
}

// ---------------------------------------------------------------------------
// Max-pool over K samples per group; write into combined feature buffer
// ---------------------------------------------------------------------------
__global__ void maxpool_kernel(const bf16* __restrict__ X, int np, int K, int coutp,
                               int coutA, bf16* __restrict__ out, int ctot, int coff) {
  int i = blockIdx.x * blockDim.x + threadIdx.x;
  if (i >= np * coutA) return;
  int g = i / coutA, c = i - g * coutA;
  float m = -1e30f;
  for (int k = 0; k < K; k++) {
    float v = (float)X[((size_t)g * K + k) * coutp + c];
    m = fmaxf(m, v);
  }
  out[(size_t)g * ctot + coff + c] = (bf16)m;
}

// ---------------------------------------------------------------------------
// 3-NN search + inverse-distance weights
// ---------------------------------------------------------------------------
__global__ void knn3_kernel(const float* __restrict__ xyz1, int N1,
                            const float* __restrict__ xyz2, int N2,
                            int* __restrict__ idx3, float* __restrict__ w3) {
  int i = blockIdx.x * blockDim.x + threadIdx.x;
  if (i >= N1) return;
  float x = xyz1[3 * i + 0], y = xyz1[3 * i + 1], z = xyz1[3 * i + 2];
  float d0 = 1e30f, d1 = 1e30f, d2 = 1e30f;
  int i0 = 0, i1 = 0, i2 = 0;
  for (int j = 0; j < N2; j++) {
    float dx = xyz2[3 * j + 0] - x;
    float dy = xyz2[3 * j + 1] - y;
    float dz = xyz2[3 * j + 2] - z;
    float d = dx * dx + dy * dy + dz * dz;
    if (d < d0)      { d2 = d1; i2 = i1; d1 = d0; i1 = i0; d0 = d; i0 = j; }
    else if (d < d1) { d2 = d1; i2 = i1; d1 = d; i1 = j; }
    else if (d < d2) { d2 = d; i2 = j; }
  }
  float w0 = 1.f / (d0 + 1e-8f);
  float w1 = 1.f / (d1 + 1e-8f);
  float w2 = 1.f / (d2 + 1e-8f);
  float s = 1.f / (w0 + w1 + w2);
  idx3[3 * i + 0] = i0; idx3[3 * i + 1] = i1; idx3[3 * i + 2] = i2;
  w3[3 * i + 0] = w0 * s; w3[3 * i + 1] = w1 * s; w3[3 * i + 2] = w2 * s;
}

// ---------------------------------------------------------------------------
// Interpolate + concat: X[n, :] = [p1[n,:C1], sum_j w*p2[idx_j, :C2], 0-pad]
// ---------------------------------------------------------------------------
__global__ void interp_concat_kernel(const bf16* __restrict__ p1, int C1,
                                     const bf16* __restrict__ p2, int C2,
                                     const int* __restrict__ idx3, const float* __restrict__ w3,
                                     int N1, int cinp, bf16* __restrict__ X) {
  size_t i = (size_t)blockIdx.x * blockDim.x + threadIdx.x;
  size_t total = (size_t)N1 * cinp;
  if (i >= total) return;
  int n = (int)(i / cinp);
  int c = (int)(i - (size_t)n * cinp);
  float v = 0.f;
  if (c < C1) {
    v = (float)p1[(size_t)n * C1 + c];
  } else if (c < C1 + C2) {
    int cc = c - C1;
    v = w3[3 * n + 0] * (float)p2[(size_t)idx3[3 * n + 0] * C2 + cc]
      + w3[3 * n + 1] * (float)p2[(size_t)idx3[3 * n + 1] * C2 + cc]
      + w3[3 * n + 2] * (float)p2[(size_t)idx3[3 * n + 2] * C2 + cc];
  }
  X[i] = (bf16)v;
}

// ===========================================================================
extern "C" void kernel_launch(void* const* d_in, const int* in_sizes, int n_in,
                              void* d_out, int out_size, void* d_ws, size_t ws_size,
                              hipStream_t stream) {
  if (n_in < 133) return;

  struct LD { int base, cin, cout; };
  static const LD L[33] = {
    /* fp1 */ {1,128,128},{5,128,128},{9,128,128},
    /* fp2 */ {13,352,256},{17,256,128},
    /* fp3 */ {21,512,256},{25,256,256},
    /* fp4 */ {29,1536,256},{33,256,256},
    /* sa1b0 */ {37,12,16},{41,16,16},{45,16,32},
    /* sa1b1 */ {49,12,32},{53,32,32},{57,32,64},
    /* sa2b0 */ {61,99,64},{65,64,64},{69,64,128},
    /* sa2b1 */ {73,99,64},{77,64,96},{81,96,128},
    /* sa3b0 */ {85,259,128},{89,128,196},{93,196,256},
    /* sa3b1 */ {97,259,128},{101,128,196},{105,196,256},
    /* sa4b0 */ {109,515,256},{113,256,256},{117,256,512},
    /* sa4b1 */ {121,515,256},{125,256,384},{129,384,512},
  };
  enum { FP1 = 0, FP2 = 3, FP3 = 5, FP4 = 7,
         SA1B0 = 9, SA1B1 = 12, SA2B0 = 15, SA2B1 = 18,
         SA3B0 = 21, SA3B1 = 24, SA4B0 = 27, SA4B1 = 30 };

  char* wsb = (char*)d_ws;
  size_t off = 0;
  auto alloc = [&](size_t bytes) -> void* {
    size_t a = (off + 255) & ~(size_t)255;
    void* r = wsb + a;
    off = a + bytes;
    return r;
  };

  // --- packed weights ---
  bf16* Wp[33]; float* Bp[33];
  for (int i = 0; i < 33; i++) {
    int cinp = ceil32(L[i].cin), coutp = ceil32(L[i].cout);
    Wp[i] = (bf16*)alloc((size_t)cinp * coutp * sizeof(bf16));
    Bp[i] = (float*)alloc((size_t)coutp * sizeof(float));
  }

  // --- pipeline buffers ---
  const int N0 = 131072;
  float* xyz0  = (float*)alloc((size_t)N0 * 3 * 4);
  bf16*  feat0 = (bf16*)alloc((size_t)N0 * 9 * 2);
  int*   fpsidx = (int*)alloc(512 * 4);
  float* xyz1 = (float*)alloc(512 * 3 * 4);
  float* xyz2 = (float*)alloc(256 * 3 * 4);
  float* xyz3 = (float*)alloc(64 * 3 * 4);
  float* xyz4 = (float*)alloc(16 * 3 * 4);
  int*   ballidx = (int*)alloc(512 * 32 * 4);
  bf16*  l1p  = (bf16*)alloc((size_t)512 * 96 * 2);
  bf16*  l2p  = (bf16*)alloc((size_t)256 * 256 * 2);
  bf16*  l3p  = (bf16*)alloc((size_t)64 * 512 * 2);
  bf16*  l4p  = (bf16*)alloc((size_t)16 * 1024 * 2);
  bf16*  l3p2 = (bf16*)alloc((size_t)64 * 256 * 2);
  bf16*  l2p2 = (bf16*)alloc((size_t)256 * 256 * 2);
  bf16*  l1p2 = (bf16*)alloc((size_t)512 * 128 * 2);
  int*   idx3 = (int*)alloc((size_t)N0 * 3 * 4);
  float* w3   = (float*)alloc((size_t)N0 * 3 * 4);
  float* pval = (float*)alloc(16 * 4);            // FPS partial argmax values
  int*   pidx = (int*)alloc(16 * 4);              // FPS partial argmax indices
  int*   sync2 = (int*)alloc(2 * 4);              // FPS arrive/gen counters
  float* gxyz  = (float*)alloc(3 * 4);            // FPS published centroid
  size_t big = (size_t)N0 * 128;                  // largest M*C (fp1)
  bf16* Xbuf = (bf16*)alloc(big * 2);
  bf16* ping = (bf16*)alloc(big * 2);
  bf16* pong = (bf16*)alloc(big * 2);
  if (off > ws_size) return;

  // --- pack all weights ---
  for (int i = 0; i < 33; i++) {
    int cinp = ceil32(L[i].cin), coutp = ceil32(L[i].cout);
    const float* b  = (const float*)d_in[L[i].base + 0];
    const float* be = (const float*)d_in[L[i].base + 1];
    const float* g  = (const float*)d_in[L[i].base + 2];
    const float* w  = (const float*)d_in[L[i].base + 3];
    int tot = cinp * coutp;
    pack_w_kernel<<<(unsigned)cdivl(tot, 256), 256, 0, stream>>>(w, g, L[i].cout, L[i].cin,
                                                                 cinp, coutp, Wp[i]);
    pack_b_kernel<<<(unsigned)cdivl(coutp, 256), 256, 0, stream>>>(b, g, be, L[i].cout,
                                                                   coutp, Bp[i]);
  }

  auto fps = [&](const float* xyzp, int N, int np, int* outidx) {
    int G = (int)cdivl(N, FPS_SLICE);
    if (G > 16) G = 16;
    if (G < 1) G = 1;
    fps_init_kernel<<<1, 1, 0, stream>>>(sync2);
    fps_multi_kernel<<<(unsigned)G, FPS_T, 0, stream>>>(xyzp, N, np, G, outidx,
                                                        pval, pidx, sync2, gxyz);
  };

  auto gemm = [&](const bf16* X, int M, int li, bf16* yb, float* yf) {
    int cinp = ceil32(L[li].cin), coutp = ceil32(L[li].cout);
    if (coutp % 64 == 0) {
      long tiles = (long)(M / 16) * (coutp / 64);
      wmma_gemm_kernel<4><<<(unsigned)cdivl(tiles, 8), 256, 0, stream>>>(
          X, Wp[li], Bp[li], M, cinp, coutp, yb, yf);
    } else {
      long tiles = (long)(M / 16) * (coutp / 32);
      wmma_gemm_kernel<2><<<(unsigned)cdivl(tiles, 8), 256, 0, stream>>>(
          X, Wp[li], Bp[li], M, cinp, coutp, yb, yf);
    }
  };

  auto run_mlp = [&](const bf16* X, int M, int start, int nl, bf16* lastB, float* lastF) {
    const bf16* cur = X;
    for (int l = 0; l < nl; l++) {
      bool last = (l == nl - 1);
      bf16* yb = last ? lastB : ((l & 1) ? pong : ping);
      float* yf = last ? lastF : nullptr;
      gemm(cur, M, start + l, yb, yf);
      cur = yb;
    }
  };

  auto sa_branch = [&](const float* xyz_in, int N, const bf16* feats, int C,
                       const float* nxyz, int np, float r, int K, int lstart,
                       int coutA, bf16* outfeat, int ctot, int coff) {
    ball_query_kernel<<<(unsigned)cdivl(np, 8), 256, 0, stream>>>(xyz_in, N, nxyz, np,
                                                                  r * r, K, ballidx);
    int cinp = ceil32(C + 3);
    sa_gather_kernel<<<(unsigned)(np * K), 64, 0, stream>>>(xyz_in, feats, C, nxyz,
                                                            ballidx, K, cinp, Xbuf);
    run_mlp(Xbuf, np * K, lstart, 3, /*lastB=*/Xbuf, nullptr);  // final -> Xbuf (input dead)
    int coutp = ceil32(coutA);
    maxpool_kernel<<<(unsigned)cdivl((long)np * coutA, 256), 256, 0, stream>>>(
        Xbuf, np, K, coutp, coutA, outfeat, ctot, coff);
  };

  auto sa_level = [&](const float* xyz_in, int N, const bf16* feats, int C,
                      int np, float* xyz_out,
                      float r0, int K0, int b0, int c0,
                      float r1, int K1, int b1, int c1,
                      bf16* outfeat, int ctot) {
    fps(xyz_in, N, np, fpsidx);
    gather_xyz_kernel<<<(unsigned)cdivl(np, 256), 256, 0, stream>>>(xyz_in, fpsidx, np,
                                                                    xyz_out);
    sa_branch(xyz_in, N, feats, C, xyz_out, np, r0, K0, b0, c0, outfeat, ctot, 0);
    sa_branch(xyz_in, N, feats, C, xyz_out, np, r1, K1, b1, c1, outfeat, ctot, c0);
  };

  auto fp_level = [&](const float* xa, int NA, const float* xb, int NB,
                      const bf16* p1, int C1, const bf16* p2, int C2,
                      int start, int nl, bf16* lastB, float* lastF) {
    knn3_kernel<<<(unsigned)cdivl(NA, 256), 256, 0, stream>>>(xa, NA, xb, NB, idx3, w3);
    int cinp = ceil32(C1 + C2);
    size_t tot = (size_t)NA * cinp;
    interp_concat_kernel<<<(unsigned)cdivl((long)tot, 256), 256, 0, stream>>>(
        p1, C1, p2, C2, idx3, w3, NA, cinp, Xbuf);
    run_mlp(Xbuf, NA, start, nl, lastB, lastF);
  };

  // --- pipeline ---
  extract_kernel<<<(unsigned)cdivl(N0, 256), 256, 0, stream>>>((const float*)d_in[0], N0,
                                                               xyz0, feat0);

  sa_level(xyz0, N0,  feat0, 9,   512, xyz1, 0.05f, 16, SA1B0, 32,  0.1f, 32, SA1B1, 64,
           l1p, 96);
  sa_level(xyz1, 512, l1p,  96,   256, xyz2, 0.1f,  16, SA2B0, 128, 0.2f, 32, SA2B1, 128,
           l2p, 256);
  sa_level(xyz2, 256, l2p, 256,    64, xyz3, 0.2f,  16, SA3B0, 256, 0.4f, 32, SA3B1, 256,
           l3p, 512);
  sa_level(xyz3, 64,  l3p, 512,    16, xyz4, 0.4f,  16, SA4B0, 512, 0.8f, 32, SA4B1, 512,
           l4p, 1024);

  fp_level(xyz3, 64,  xyz4, 16,  l3p, 512,  l4p, 1024, FP4, 2, l3p2, nullptr);
  fp_level(xyz2, 256, xyz3, 64,  l2p, 256,  l3p2, 256, FP3, 2, l2p2, nullptr);
  fp_level(xyz1, 512, xyz2, 256, l1p, 96,   l2p2, 256, FP2, 2, l1p2, nullptr);
  fp_level(xyz0, N0,  xyz1, 512, nullptr, 0, l1p2, 128, FP1, 3, nullptr, (float*)d_out);

  (void)in_sizes; (void)out_size;
}
